// MultiHeadSelfAttention_41893111005251
// MI455X (gfx1250) — matrix operand order repaired and checked by reading, not compiled, not run on hardware
//
#include <hip/hip_runtime.h>
#include <hip/hip_bf16.h>
#include <math.h>
#include <stdint.h>

typedef __attribute__((ext_vector_type(16))) _Float16 v16h;
typedef __attribute__((ext_vector_type(8)))  _Float16 v8h;
typedef __attribute__((ext_vector_type(8)))  float    v8f;

static constexpr int BATCH  = 2;
static constexpr int SEQ    = 2048;
static constexpr int TOKS   = BATCH * SEQ;   // 4096
static constexpr int DMODEL = 768;
static constexpr int NHEAD  = 12;
static constexpr int DHEAD  = 64;
#define L2E 1.44269504088896340736f

__device__ __forceinline__ v16h cat8(v8h lo, v8h hi) {
  return __builtin_shufflevector(lo, hi, 0,1,2,3,4,5,6,7,8,9,10,11,12,13,14,15);
}

__device__ __forceinline__ v8f wmma_f16(v16h a, v16h b, v8f c) {
  // (neg_a, A, neg_b, B, c_mod, C, reuse_a, reuse_b)
  return __builtin_amdgcn_wmma_f32_16x16x32_f16(false, a, false, b, (short)0, c, false, false);
}

// ---- CDNA5 async global->LDS copy (ASYNCcnt-tracked), GV addressing --------
// vdst VGPR = wave-relative LDS byte address; vaddr = 64-bit global address.
__device__ __forceinline__ void async_b128(uint32_t lds_byte_off, const void* gptr) {
  asm volatile("global_load_async_to_lds_b128 %0, %1, off"
               :: "v"(lds_byte_off), "v"((uint64_t)(uintptr_t)gptr)
               : "memory");
}
template <int N>
__device__ __forceinline__ void wait_asynccnt() {
  asm volatile("s_wait_asynccnt %0" :: "i"(N) : "memory");
}
// flat shared address -> wave-relative LDS byte offset (aperture: low 32 bits)
__device__ __forceinline__ uint32_t lds_off(const void* p) {
  return (uint32_t)(uintptr_t)p;
}

// ---------------------------------------------------------------- conversions
__global__ void cast_f16(const float* __restrict__ in, _Float16* __restrict__ out, int n) {
  int i = blockIdx.x * blockDim.x + threadIdx.x;
  if (i < n) out[i] = (_Float16)in[i];
}

// W[k][n] (f32, KxN) -> Wt[n][k] (f16, NxK) so WMMA B-fragments are K-contiguous
__global__ void transpose_cast_f16(const float* __restrict__ W, _Float16* __restrict__ Wt,
                                   int K, int N) {
  int i = blockIdx.x * blockDim.x + threadIdx.x;
  if (i < K * N) {
    int k = i / N;
    int n = i - k * N;
    Wt[(size_t)n * K + k] = (_Float16)W[i];
  }
}

// ---------------------------------------------------------------- WMMA GEMM
// Y[M,N] = X[M,K] @ W[K,N] + bias   (W given transposed: Wt[N][K], f16)
// Block = 64x64 output tile: 4 waves (M subtiles) share one B tile staged in
// LDS by double-buffered async copies (each thread stages 2x16B per tile).
// MODE 0: f16 out, 1: f16 transposed out, 2: f32 out.
template <int MODE>
__global__ __launch_bounds__(128)
void gemm_wmma(const _Float16* __restrict__ X, const _Float16* __restrict__ Wt,
               const float* __restrict__ bias, void* __restrict__ Yv,
               int M, int N, int Kd) {
  __shared__ __align__(32) _Float16 bbuf[2][64 * 32];   // [n_local][k_local]
  const int tid  = threadIdx.x;
  const int lane = tid & 31;
  const int wv   = tid >> 5;
  const int n16  = lane & 15;
  const int hg   = lane >> 4;          // half-wave group
  const int ntiles = N >> 6;           // 64-wide N tiles
  const int mt64 = blockIdx.x / ntiles;
  const int nt   = blockIdx.x - mt64 * ntiles;
  const int n0   = nt * 64;
  const int m0   = mt64 * 64 + wv * 16;   // this wave's 16 rows

  v8f acc[4];
#pragma unroll
  for (int c = 0; c < 4; ++c) {
    float bb = bias[n0 + c * 16 + n16];
#pragma unroll
    for (int r = 0; r < 8; ++r) acc[c][r] = bb;
  }

  const _Float16* xrow = X + (size_t)(m0 + n16) * Kd + 16 * hg;
  // staging: thread t copies 32B of the 64x32 B tile (row = t>>1, half = t&1)
  const int srow  = tid >> 1;
  const int shalf = tid & 1;
  const _Float16* gsrc = Wt + (size_t)(n0 + srow) * Kd + shalf * 16;
  const uint32_t  ldst = lds_off(&bbuf[0][0]) + (uint32_t)(srow * 32 + shalf * 16) * 2u;
  const uint32_t  bstride = 64u * 32u * 2u;

  // preload tile 0: this thread's two 16B pieces
  async_b128(ldst, gsrc);
  async_b128(ldst + 16u, (const char*)gsrc + 16);

  const int iters = Kd >> 5;   // even (Kd = 768 -> 24)
#pragma unroll 2
  for (int it = 0; it < iters; ++it) {
    const int kb  = it << 5;
    const int buf = it & 1;
    if (it + 1 < iters) {
      const _Float16* gn = gsrc + kb + 32;
      const uint32_t  ln = ldst + (uint32_t)((it + 1) & 1) * bstride;
      async_b128(ln, gn);
      async_b128(ln + 16u, (const char*)gn + 16);
      wait_asynccnt<2>();   // oldest 2 (current tile) complete
    } else {
      wait_asynccnt<0>();
    }
    __syncthreads();

    v16h a = cat8(*(const v8h*)(xrow + kb), *(const v8h*)(xrow + kb + 8));
    const _Float16* bb = &bbuf[buf][0] + 16 * hg;
#pragma unroll
    for (int c = 0; c < 4; ++c) {
      v16h bf = *(const v16h*)(bb + (c * 16 + n16) * 32);
      acc[c] = wmma_f16(a, bf, acc[c]);
    }
    __builtin_prefetch(xrow + kb + 32, 0, 0);   // global_prefetch_b8 (A stream)
    __syncthreads();   // all waves done reading buf before it is overwritten
  }

  if constexpr (MODE == 0) {
    _Float16* Y = (_Float16*)Yv;
#pragma unroll
    for (int c = 0; c < 4; ++c)
#pragma unroll
      for (int r = 0; r < 8; ++r)
        Y[(size_t)(m0 + r + 8 * hg) * N + n0 + c * 16 + n16] = (_Float16)acc[c][r];
  } else if constexpr (MODE == 1) {
    _Float16* Y = (_Float16*)Yv;
#pragma unroll
    for (int c = 0; c < 4; ++c) {
      v8h hv = {};
#pragma unroll
      for (int r = 0; r < 8; ++r) hv[r] = (_Float16)acc[c][r];
      *(v8h*)(Y + (size_t)(n0 + c * 16 + n16) * M + m0 + 8 * hg) = hv;
    }
  } else {
    float* Y = (float*)Yv;
#pragma unroll
    for (int c = 0; c < 4; ++c)
#pragma unroll
      for (int r = 0; r < 8; ++r)
        Y[(size_t)(m0 + r + 8 * hg) * N + n0 + c * 16 + n16] = acc[c][r];
  }
}

// ---------------------------------------------------------------- flash attention
// Q,K: [TOKS][DMODEL] f16 ; Vt: [DMODEL][TOKS] f16 ; Ctx: [TOKS][DMODEL] f16
// Block = 4 q-tiles of the SAME (batch, head): K/V tiles staged once per block
// into double-buffered LDS with async copies (4x16B per thread per tile).
__global__ __launch_bounds__(128)
void flash_attn(const _Float16* __restrict__ Q, const _Float16* __restrict__ Kk,
                const _Float16* __restrict__ Vt, _Float16* __restrict__ Ctx) {
  __shared__ __align__(32) _Float16 ktile[2][32 * 64];  // [key][dh]
  __shared__ __align__(32) _Float16 vtile[2][64 * 32];  // [dh][key]
  __shared__ __align__(32) _Float16 pbuf[4][16 * 32];
  const int tid  = threadIdx.x;
  const int lane = tid & 31;
  const int wv   = tid >> 5;
  const int n16  = lane & 15;
  const int hg   = lane >> 4;
  const int wid  = blockIdx.x * 4 + wv;          // exact grid: no early exit
  const int qt = wid & 127;
  const int h  = (wid >> 7) % NHEAD;
  const int b  = wid / (128 * NHEAD);
  const int tokq  = b * SEQ + qt * 16;
  const int tokk0 = b * SEQ;

  // Q tile 16x64 -> two A fragments (Dh chunks of 32)
  const _Float16* qp = Q + (size_t)(tokq + n16) * DMODEL + h * DHEAD + 16 * hg;
  v16h a0 = cat8(*(const v8h*)(qp),      *(const v8h*)(qp + 8));
  v16h a1 = cat8(*(const v8h*)(qp + 32), *(const v8h*)(qp + 40));

  v16h ones;
#pragma unroll
  for (int e = 0; e < 16; ++e) ones[e] = (_Float16)1.0f;

  // staging assignments (whole block covers each 4KB tile; 32B per thread)
  const int krow = tid >> 2, kchunk = tid & 3;          // K: 32 rows x 4 chunks
  const int vrow = tid >> 1, vhalf  = tid & 1;          // V: 64 rows x 2 halves
  const _Float16* kgsrc = Kk + (size_t)(tokk0 + krow) * DMODEL + h * DHEAD + kchunk * 16;
  const _Float16* vgsrc = Vt + (size_t)(h * DHEAD + vrow) * TOKS + tokk0 + vhalf * 16;
  const uint32_t klds = lds_off(&ktile[0][0]) + (uint32_t)(krow * 64 + kchunk * 16) * 2u;
  const uint32_t vlds = lds_off(&vtile[0][0]) + (uint32_t)(vrow * 32 + vhalf * 16) * 2u;
  const uint32_t kstride = 32u * 64u * 2u;
  const uint32_t vstride = 64u * 32u * 2u;

  float mhalf = -3.0e38f;
  float lrow[8];
  v8f acc[4] = {{}, {}, {}, {}};
#pragma unroll
  for (int r = 0; r < 8; ++r) lrow[r] = 0.f;
  _Float16* pb = pbuf[wv];

  // preload tile 0 (4 async b128 per thread)
  async_b128(klds, kgsrc);
  async_b128(klds + 32u, (const char*)kgsrc + 32);
  async_b128(vlds, vgsrc);
  async_b128(vlds + 32u, (const char*)vgsrc + 32);

  const int iters = SEQ / 32;   // 64 (even)
#pragma unroll 2
  for (int it = 0; it < iters; ++it) {
    const int kb  = it << 5;
    const int buf = it & 1;
    if (it + 1 < iters) {
      const uint32_t kl = klds + (uint32_t)((it + 1) & 1) * kstride;
      const uint32_t vl = vlds + (uint32_t)((it + 1) & 1) * vstride;
      const _Float16* kg = kgsrc + (size_t)(kb + 32) * DMODEL;
      const _Float16* vg = vgsrc + kb + 32;
      async_b128(kl, kg);
      async_b128(kl + 32u, (const char*)kg + 32);
      async_b128(vl, vg);
      async_b128(vl + 32u, (const char*)vg + 32);
      wait_asynccnt<4>();   // oldest 4 (current tile) complete
    } else {
      wait_asynccnt<0>();
    }
    __syncthreads();

    // scores: two 16x16 tiles (key subtiles) over Dh=64, fragments from LDS
    const _Float16* kt = &ktile[buf][0] + n16 * 64 + 16 * hg;
    v16h bk00 = *(const v16h*)(kt);
    v16h bk01 = *(const v16h*)(kt + 32);
    v16h bk10 = *(const v16h*)(kt + 16 * 64);
    v16h bk11 = *(const v16h*)(kt + 16 * 64 + 32);
    v8f s0 = {}, s1 = {};
    s0 = wmma_f16(a0, bk00, s0);  s0 = wmma_f16(a1, bk01, s0);
    s1 = wmma_f16(a0, bk10, s1);  s1 = wmma_f16(a1, bk11, s1);

    // half-tile running max: in-lane tree + 4 xor-shuffles over 16 lanes
    float t0 = fmaxf(fmaxf(s0[0], s0[1]), fmaxf(s0[2], s0[3]));
    float t1 = fmaxf(fmaxf(s0[4], s0[5]), fmaxf(s0[6], s0[7]));
    float t2 = fmaxf(fmaxf(s1[0], s1[1]), fmaxf(s1[2], s1[3]));
    float t3 = fmaxf(fmaxf(s1[4], s1[5]), fmaxf(s1[6], s1[7]));
    float mt = fmaxf(fmaxf(t0, t1), fmaxf(t2, t3));
    mt = fmaxf(mt, __shfl_xor(mt, 1, 32));
    mt = fmaxf(mt, __shfl_xor(mt, 2, 32));
    mt = fmaxf(mt, __shfl_xor(mt, 4, 32));
    mt = fmaxf(mt, __shfl_xor(mt, 8, 32));

    const float mn   = fmaxf(mhalf, mt);
    const float corr = exp2f((mhalf - mn) * L2E);
    mhalf = mn;

#pragma unroll
    for (int r = 0; r < 8; ++r) {
      s0[r] = exp2f((s0[r] - mn) * L2E);
      s1[r] = exp2f((s1[r] - mn) * L2E);
      lrow[r] *= corr;
    }
#pragma unroll
    for (int c = 0; c < 4; ++c)
#pragma unroll
      for (int r = 0; r < 8; ++r) acc[c][r] *= corr;

    // P (16x32, C layout) -> LDS row-major -> A fragment
#pragma unroll
    for (int r = 0; r < 8; ++r) {
      pb[(r + 8 * hg) * 32 + n16]      = (_Float16)s0[r];
      pb[(r + 8 * hg) * 32 + 16 + n16] = (_Float16)s1[r];
    }
    __builtin_amdgcn_wave_barrier();
    const _Float16* prow = pb + n16 * 32 + 16 * hg;
    v16h pa = cat8(*(const v8h*)(prow), *(const v8h*)(prow + 8));

    // exact per-row sums: P @ ones
    v8f rs = {};
    rs = wmma_f16(pa, ones, rs);
#pragma unroll
    for (int r = 0; r < 8; ++r) lrow[r] += rs[r];

    // ctx += P @ V  (V tile in LDS, [dh][key])
    const _Float16* vt = &vtile[buf][0] + n16 * 32 + 16 * hg;
#pragma unroll
    for (int c = 0; c < 4; ++c) {
      v16h bv = *(const v16h*)(vt + c * 16 * 32);
      acc[c] = wmma_f16(pa, bv, acc[c]);
    }
    __syncthreads();   // all waves done with buf before it is overwritten
  }

  float invl[8];
#pragma unroll
  for (int r = 0; r < 8; ++r) invl[r] = 1.0f / lrow[r];
#pragma unroll
  for (int c = 0; c < 4; ++c)
#pragma unroll
    for (int r = 0; r < 8; ++r)
      Ctx[(size_t)(tokq + r + 8 * hg) * DMODEL + h * DHEAD + c * 16 + n16] =
          (_Float16)(acc[c][r] * invl[r]);
}

// ---------------------------------------------------------------- launcher
extern "C" void kernel_launch(void* const* d_in, const int* in_sizes, int n_in,
                              void* d_out, int out_size, void* d_ws, size_t ws_size,
                              hipStream_t stream) {
  const float* x  = (const float*)d_in[0];
  const float* Wq = (const float*)d_in[1];
  const float* bq = (const float*)d_in[2];
  const float* Wk = (const float*)d_in[3];
  const float* bk = (const float*)d_in[4];
  const float* Wv = (const float*)d_in[5];
  const float* bv = (const float*)d_in[6];
  const float* Wo = (const float*)d_in[7];
  const float* bo = (const float*)d_in[8];

  const size_t XSZ = (size_t)TOKS * DMODEL;     // 3,145,728 halves
  const size_t WSZ = (size_t)DMODEL * DMODEL;   //   589,824 halves

  _Float16* x16 = (_Float16*)d_ws;          // reused as ctx after V GEMM
  _Float16* wqT = x16 + XSZ;
  _Float16* wkT = wqT + WSZ;
  _Float16* wvT = wkT + WSZ;
  _Float16* woT = wvT + WSZ;
  _Float16* q16 = woT + WSZ;
  _Float16* k16 = q16 + XSZ;
  _Float16* vT  = k16 + XSZ;
  _Float16* ctx = x16;                      // alias: x16 dead after QKV GEMMs

  cast_f16<<<(int)((XSZ + 255) / 256), 256, 0, stream>>>(x, x16, (int)XSZ);
  transpose_cast_f16<<<(int)((WSZ + 255) / 256), 256, 0, stream>>>(Wq, wqT, DMODEL, DMODEL);
  transpose_cast_f16<<<(int)((WSZ + 255) / 256), 256, 0, stream>>>(Wk, wkT, DMODEL, DMODEL);
  transpose_cast_f16<<<(int)((WSZ + 255) / 256), 256, 0, stream>>>(Wv, wvT, DMODEL, DMODEL);
  transpose_cast_f16<<<(int)((WSZ + 255) / 256), 256, 0, stream>>>(Wo, woT, DMODEL, DMODEL);

  const int blocks = (TOKS / 64) * (DMODEL / 64);   // 64x64 tile per block
  gemm_wmma<0><<<blocks, 128, 0, stream>>>(x16, wqT, bq, q16, TOKS, DMODEL, DMODEL);
  gemm_wmma<0><<<blocks, 128, 0, stream>>>(x16, wkT, bk, k16, TOKS, DMODEL, DMODEL);
  gemm_wmma<1><<<blocks, 128, 0, stream>>>(x16, wvT, bv, vT,  TOKS, DMODEL, DMODEL);

  const int awaves = BATCH * NHEAD * (SEQ / 16);    // 3072 (exact grid)
  flash_attn<<<awaves / 4, 128, 0, stream>>>(q16, k16, vT, ctx);

  gemm_wmma<2><<<blocks, 128, 0, stream>>>(ctx, woT, bo, d_out, TOKS, DMODEL, DMODEL);

  (void)in_sizes; (void)n_in; (void)out_size; (void)ws_size;
}
